// SpatialGNN_20486994002019
// MI455X (gfx1250) — matrix operand order repaired and checked by reading, not compiled, not run on hardware
//
#include <hip/hip_runtime.h>

// ---------------------------------------------------------------------------
// CDNA5 (gfx1250) implementation of the SpatialGNN reference.
// GEMMs run on the WMMA unit via split-precision bf16 (hi/lo) with fp32
// accumulate: A*B ~= Ah*Bh + Al*Bh + Ah*Bl  -> ~fp32 accuracy, 3 v_wmma per
// 16x16x32 K-step. Each wave computes a 16x32 C tile (two 16x16 WMMA tiles)
// so the A hi/lo fragments are reused across two B fragments; K is a template
// parameter so the K-loop fully unrolls. GCN scatter uses L2-resident fp32
// atomics. ICP+chamfer is fused: one workgroup per graph, clouds held in LDS,
// deterministic wave32 shuffle reductions, 3x3 Jacobi SVD on one lane.
// ---------------------------------------------------------------------------

typedef __attribute__((ext_vector_type(16))) __bf16 v16bf;
typedef __attribute__((ext_vector_type(8)))  float  v8f;

namespace {
constexpr int kN    = 16384;
constexpr int kB    = 8;
constexpr int kNPG  = kN / kB;     // 2048
constexpr int kE    = 262144;
constexpr int kIN   = 128;
constexpr int kHID  = 256;
constexpr int kOUT  = 128;
constexpr int kIcpIters = 10;
constexpr int kIcpThreads = 1024;  // 32 waves
}

__device__ __forceinline__ v8f wmma_bf16(const v16bf& a, const v16bf& b,
                                         const v8f& c) {
  return __builtin_amdgcn_wmma_f32_16x16x32_bf16(false, a, false, b,
                                                 (short)0, c, false, false);
}

// ------------------------------ WMMA GEMM ----------------------------------
// C[M,N] = A[M,K] @ B[K,N], fp32 row-major, M%16==0, N%32==0, K%32==0.
// One wave computes a 16x32 C tile. A/B fragments use contiguous k = 16*half + e (same order in both).
template <int K>
__global__ void __launch_bounds__(256)
gemm_wmma_bf16x3(const float* __restrict__ A, const float* __restrict__ B,
                 float* __restrict__ C, int M, int N) {
  const int lane = threadIdx.x & 31;
  const int wid  = blockIdx.x * (blockDim.x >> 5) + (threadIdx.x >> 5);
  const int tilesN = N >> 5;              // 32-wide wave tiles
  const int tm = (wid / tilesN) << 4;
  const int tn = (wid % tilesN) << 5;
  if (tm >= M) return;
  const int half = lane >> 4;             // lanes 16..31 hold second K-half
  const int l16  = lane & 15;

  v8f acc0 = {}, acc1 = {};
#pragma unroll
  for (int k0 = 0; k0 < K; k0 += 32) {
    v16bf ah, al, b0h, b0l, b1h, b1l;
    // A fragment: lane l16 = row M, element e -> K = 16*half + e
    const float* arow = A + (size_t)(tm + l16) * K + k0;
#pragma unroll
    for (int e = 0; e < 16; ++e) {
      const int kk = (half << 4) | e;
      const float v = arow[kk];
      const __bf16 hi = (__bf16)v;
      ah[e] = hi;
      al[e] = (__bf16)(v - (float)hi);
    }
    // B fragments: lanes 0-15 K=0..15, lanes 16-31 K=16..31; N = l16 (+16)
#pragma unroll
    for (int e = 0; e < 16; ++e) {
      const int kk = (half << 4) | e;
      const float* brow = B + (size_t)(k0 + kk) * N + tn + l16;
      const float v0 = brow[0];
      const float v1 = brow[16];
      const __bf16 h0 = (__bf16)v0;
      const __bf16 h1 = (__bf16)v1;
      b0h[e] = h0; b0l[e] = (__bf16)(v0 - (float)h0);
      b1h[e] = h1; b1l[e] = (__bf16)(v1 - (float)h1);
    }
    acc0 = wmma_bf16(ah, b0h, acc0);
    acc0 = wmma_bf16(al, b0h, acc0);
    acc0 = wmma_bf16(ah, b0l, acc0);
    acc1 = wmma_bf16(ah, b1h, acc1);
    acc1 = wmma_bf16(al, b1h, acc1);
    acc1 = wmma_bf16(ah, b1l, acc1);
  }
  // C/D layout: VGPR i -> row tm + i + 8*half, col tn + l16 (+16)
  float* crow = C + (size_t)(tm + (half << 3)) * N + tn + l16;
#pragma unroll
  for (int i = 0; i < 8; ++i) {
    crow[(size_t)i * N]      = acc0[i];
    crow[(size_t)i * N + 16] = acc1[i];
  }
}

// ------------------------------ GCN pieces ---------------------------------
__global__ void deg_init_kernel(float* __restrict__ deg, int n) {
  int t = blockIdx.x * blockDim.x + threadIdx.x;
  if (t < n) deg[t] = 1.0f;           // self loop
}

__global__ void deg_edge_kernel(const int* __restrict__ col,
                                float* __restrict__ deg, int e) {
  int t = blockIdx.x * blockDim.x + threadIdx.x;
  if (t < e) atomicAdd(&deg[col[t]], 1.0f);   // exact integer fp adds
}

__global__ void dinv_kernel(float* __restrict__ deg, int n) {
  int t = blockIdx.x * blockDim.x + threadIdx.x;
  if (t < n) deg[t] = rsqrtf(deg[t]);          // deg >= 1 always
}

// acc[col] += xw[row] * dinv[row]*dinv[col]; 4 features per thread
__global__ void gcn_scatter_kernel(const float* __restrict__ xw,
                                   float* __restrict__ acc,
                                   const int* __restrict__ row,
                                   const int* __restrict__ col,
                                   const float* __restrict__ dinv,
                                   int e, int F) {
  const int per = F >> 2;
  int t = blockIdx.x * blockDim.x + threadIdx.x;
  int ei = t / per;
  if (ei >= e) return;
  int f = (t - ei * per) << 2;
  int r = row[ei], c = col[ei];
  float nrm = dinv[r] * dinv[c];
  const float* s = xw + (size_t)r * F + f;
  float* d = acc + (size_t)c * F + f;
  atomicAdd(&d[0], s[0] * nrm);
  atomicAdd(&d[1], s[1] * nrm);
  atomicAdd(&d[2], s[2] * nrm);
  atomicAdd(&d[3], s[3] * nrm);
}

// out = [relu]( acc + xw*dinv^2 (self loop) + bias )
__global__ void gcn_finalize_kernel(const float* __restrict__ xw,
                                    const float* __restrict__ acc,
                                    const float* __restrict__ bias,
                                    const float* __restrict__ dinv,
                                    float* __restrict__ out,
                                    int n, int F, int relu) {
  int t = blockIdx.x * blockDim.x + threadIdx.x;
  if (t >= n * F) return;
  int node = t / F, f = t - node * F;
  float dv = dinv[node];
  float v = acc[t] + xw[t] * dv * dv + bias[f];
  out[t] = relu ? fmaxf(v, 0.0f) : v;
}

// tpos = pos + h @ Wt + bt   (Wt: [128,3])
__global__ void head_kernel(const float* __restrict__ h,
                            const float* __restrict__ Wt,
                            const float* __restrict__ bt,
                            const float* __restrict__ pos,
                            float* __restrict__ tpos, int n) {
  int t = blockIdx.x * blockDim.x + threadIdx.x;
  if (t >= n * 3) return;
  int node = t / 3, d = t - node * 3;
  float s = bt[d];
  const float* hr = h + (size_t)node * kOUT;
#pragma unroll 8
  for (int k = 0; k < kOUT; ++k) s += hr[k] * Wt[k * 3 + d];
  tpos[t] = pos[t] + s;
}

// ------------------------------ ICP + chamfer ------------------------------
__device__ __forceinline__ float wave_sum32(float v) {
#pragma unroll
  for (int off = 16; off > 0; off >>= 1) v += __shfl_down(v, off, 32);
  return v;
}

__global__ void __launch_bounds__(kIcpThreads)
icp_chamfer_kernel(const float* __restrict__ tpos, const float* __restrict__ pos,
                   float* __restrict__ alignedOut, float* __restrict__ cham) {
  __shared__ float sx[kNPG], sy[kNPG], sz[kNPG];
  __shared__ float txs[kNPG], tys[kNPG], tzs[kNPG];
  __shared__ float parts[15][32];
  __shared__ float red[16];
  __shared__ float RT[12];

  const int g = blockIdx.x;
  const int tid = threadIdx.x;
  const int wv = tid >> 5, ln = tid & 31;
  const float* sp = tpos + (size_t)g * kNPG * 3;
  const float* tp = pos  + (size_t)g * kNPG * 3;

  for (int i = tid; i < kNPG; i += kIcpThreads) {
    sx[i] = sp[3 * i];  sy[i] = sp[3 * i + 1];  sz[i] = sp[3 * i + 2];
    txs[i] = tp[3 * i]; tys[i] = tp[3 * i + 1]; tzs[i] = tp[3 * i + 2];
  }
  __syncthreads();

  for (int iter = 0; iter < kIcpIters; ++iter) {
    float lS0 = 0, lS1 = 0, lS2 = 0, lC0 = 0, lC1 = 0, lC2 = 0;
    float lH[9] = {0, 0, 0, 0, 0, 0, 0, 0, 0};
    for (int i = tid; i < kNPG; i += kIcpThreads) {
      float px = sx[i], py = sy[i], pz = sz[i];
      float best = 3.4e38f; int bj = 0;
      for (int j = 0; j < kNPG; ++j) {
        float dx = px - txs[j], dy = py - tys[j], dz = pz - tzs[j];
        float d = dx * dx + dy * dy + dz * dz;
        if (d < best) { best = d; bj = j; }
      }
      float cxv = txs[bj], cyv = tys[bj], czv = tzs[bj];
      lS0 += px; lS1 += py; lS2 += pz;
      lC0 += cxv; lC1 += cyv; lC2 += czv;
      lH[0] += px * cxv; lH[1] += px * cyv; lH[2] += px * czv;
      lH[3] += py * cxv; lH[4] += py * cyv; lH[5] += py * czv;
      lH[6] += pz * cxv; lH[7] += pz * cyv; lH[8] += pz * czv;
    }
    float vals[15] = {lS0, lS1, lS2, lC0, lC1, lC2,
                      lH[0], lH[1], lH[2], lH[3], lH[4],
                      lH[5], lH[6], lH[7], lH[8]};
#pragma unroll
    for (int q = 0; q < 15; ++q) {
      float w = wave_sum32(vals[q]);
      if (ln == 0) parts[q][wv] = w;
    }
    __syncthreads();
    if (tid < 15) {
      float s = 0.0f;
#pragma unroll
      for (int w2 = 0; w2 < 32; ++w2) s += parts[tid][w2];
      red[tid] = s;
    }
    __syncthreads();

    if (tid == 0) {
      const float invn = 1.0f / (float)kNPG;
      float cs0 = red[0] * invn, cs1 = red[1] * invn, cs2 = red[2] * invn;
      float cc0 = red[3] * invn, cc1 = red[4] * invn, cc2 = red[5] * invn;
      float csv[3] = {cs0, cs1, cs2}, ccv[3] = {cc0, cc1, cc2};
      float Hm[3][3];
#pragma unroll
      for (int a = 0; a < 3; ++a)
#pragma unroll
        for (int b = 0; b < 3; ++b)
          Hm[a][b] = red[6 + a * 3 + b] - (float)kNPG * csv[a] * ccv[b];
      // A = H^T H (symmetric)
      float Am[3][3];
#pragma unroll
      for (int i2 = 0; i2 < 3; ++i2)
#pragma unroll
        for (int j2 = 0; j2 < 3; ++j2)
          Am[i2][j2] = Hm[0][i2] * Hm[0][j2] + Hm[1][i2] * Hm[1][j2] +
                       Hm[2][i2] * Hm[2][j2];
      float V[3][3] = {{1, 0, 0}, {0, 1, 0}, {0, 0, 1}};
      for (int sweep = 0; sweep < 10; ++sweep) {
        for (int pp = 0; pp < 2; ++pp) {
          for (int qq = pp + 1; qq < 3; ++qq) {
            float apq = Am[pp][qq];
            if (fabsf(apq) < 1e-30f) continue;
            float theta = (Am[qq][qq] - Am[pp][pp]) / (2.0f * apq);
            float tt = ((theta >= 0.0f) ? 1.0f : -1.0f) /
                       (fabsf(theta) + sqrtf(theta * theta + 1.0f));
            float c = rsqrtf(tt * tt + 1.0f);
            float s = tt * c;
            Am[pp][pp] -= tt * apq;
            Am[qq][qq] += tt * apq;
            Am[pp][qq] = Am[qq][pp] = 0.0f;
            int rr = 3 - pp - qq;
            float arp = Am[rr][pp], arq = Am[rr][qq];
            Am[rr][pp] = Am[pp][rr] = c * arp - s * arq;
            Am[rr][qq] = Am[qq][rr] = s * arp + c * arq;
#pragma unroll
            for (int r2 = 0; r2 < 3; ++r2) {
              float vrp = V[r2][pp], vrq = V[r2][qq];
              V[r2][pp] = c * vrp - s * vrq;
              V[r2][qq] = s * vrp + c * vrq;
            }
          }
        }
      }
      float lam[3] = {Am[0][0], Am[1][1], Am[2][2]};
      int ord[3] = {0, 1, 2};
      if (lam[ord[0]] < lam[ord[1]]) { int t2 = ord[0]; ord[0] = ord[1]; ord[1] = t2; }
      if (lam[ord[1]] < lam[ord[2]]) { int t2 = ord[1]; ord[1] = ord[2]; ord[2] = t2; }
      if (lam[ord[0]] < lam[ord[1]]) { int t2 = ord[0]; ord[0] = ord[1]; ord[1] = t2; }
      float Vs[3][3], sig[3];
#pragma unroll
      for (int k2 = 0; k2 < 3; ++k2) {
        sig[k2] = sqrtf(fmaxf(lam[ord[k2]], 0.0f));
#pragma unroll
        for (int r2 = 0; r2 < 3; ++r2) Vs[r2][k2] = V[r2][ord[k2]];
      }
      float U[3][3];
#pragma unroll
      for (int k2 = 0; k2 < 3; ++k2) {
        float ux = Hm[0][0] * Vs[0][k2] + Hm[0][1] * Vs[1][k2] + Hm[0][2] * Vs[2][k2];
        float uy = Hm[1][0] * Vs[0][k2] + Hm[1][1] * Vs[1][k2] + Hm[1][2] * Vs[2][k2];
        float uz = Hm[2][0] * Vs[0][k2] + Hm[2][1] * Vs[1][k2] + Hm[2][2] * Vs[2][k2];
        float nrm = sqrtf(ux * ux + uy * uy + uz * uz);
        if (sig[k2] > 1e-12f && nrm > 1e-20f) {
          float inv = 1.0f / nrm;
          U[0][k2] = ux * inv; U[1][k2] = uy * inv; U[2][k2] = uz * inv;
        } else {
          float ax = U[0][0], ay = U[1][0], az = U[2][0];
          float bx = U[0][1], by = U[1][1], bz = U[2][1];
          U[0][k2] = ay * bz - az * by;
          U[1][k2] = az * bx - ax * bz;
          U[2][k2] = ax * by - ay * bx;
        }
      }
      float detU = U[0][0] * (U[1][1] * U[2][2] - U[1][2] * U[2][1]) -
                   U[0][1] * (U[1][0] * U[2][2] - U[1][2] * U[2][0]) +
                   U[0][2] * (U[1][0] * U[2][1] - U[1][1] * U[2][0]);
      float detV = Vs[0][0] * (Vs[1][1] * Vs[2][2] - Vs[1][2] * Vs[2][1]) -
                   Vs[0][1] * (Vs[1][0] * Vs[2][2] - Vs[1][2] * Vs[2][0]) +
                   Vs[0][2] * (Vs[1][0] * Vs[2][1] - Vs[1][1] * Vs[2][0]);
      float sgn = (detU * detV < 0.0f) ? -1.0f : 1.0f;
      float Rm[3][3];
#pragma unroll
      for (int i2 = 0; i2 < 3; ++i2)
#pragma unroll
        for (int j2 = 0; j2 < 3; ++j2)
          Rm[i2][j2] = Vs[i2][0] * U[j2][0] + Vs[i2][1] * U[j2][1] +
                       sgn * Vs[i2][2] * U[j2][2];
      RT[0] = Rm[0][0]; RT[1] = Rm[0][1]; RT[2] = Rm[0][2];
      RT[3] = Rm[1][0]; RT[4] = Rm[1][1]; RT[5] = Rm[1][2];
      RT[6] = Rm[2][0]; RT[7] = Rm[2][1]; RT[8] = Rm[2][2];
      RT[9]  = cc0 - (Rm[0][0] * cs0 + Rm[0][1] * cs1 + Rm[0][2] * cs2);
      RT[10] = cc1 - (Rm[1][0] * cs0 + Rm[1][1] * cs1 + Rm[1][2] * cs2);
      RT[11] = cc2 - (Rm[2][0] * cs0 + Rm[2][1] * cs1 + Rm[2][2] * cs2);
    }
    __syncthreads();
    for (int i = tid; i < kNPG; i += kIcpThreads) {
      float px = sx[i], py = sy[i], pz = sz[i];
      float nx = RT[0] * px + RT[1] * py + RT[2] * pz + RT[9];
      float ny = RT[3] * px + RT[4] * py + RT[5] * pz + RT[10];
      float nz = RT[6] * px + RT[7] * py + RT[8] * pz + RT[11];
      sx[i] = nx; sy[i] = ny; sz[i] = nz;
    }
    __syncthreads();
  }

  // Chamfer (both directions) + write aligned points
  float a1 = 0.0f, a2 = 0.0f;
  for (int i = tid; i < kNPG; i += kIcpThreads) {
    float px = sx[i], py = sy[i], pz = sz[i];
    float* ao = alignedOut + (size_t)g * kNPG * 3 + 3 * i;
    ao[0] = px; ao[1] = py; ao[2] = pz;
    float best = 3.4e38f;
    for (int j = 0; j < kNPG; ++j) {
      float dx = px - txs[j], dy = py - tys[j], dz = pz - tzs[j];
      best = fminf(best, dx * dx + dy * dy + dz * dz);
    }
    a1 += best;
    float qx = txs[i], qy = tys[i], qz = tzs[i];
    float best2 = 3.4e38f;
    for (int j = 0; j < kNPG; ++j) {
      float dx = qx - sx[j], dy = qy - sy[j], dz = qz - sz[j];
      best2 = fminf(best2, dx * dx + dy * dy + dz * dz);
    }
    a2 += best2;
  }
  float w1 = wave_sum32(a1), w2 = wave_sum32(a2);
  if (ln == 0) { parts[0][wv] = w1; parts[1][wv] = w2; }
  __syncthreads();
  if (tid == 0) {
    float s1 = 0.0f, s2 = 0.0f;
#pragma unroll
    for (int w = 0; w < 32; ++w) { s1 += parts[0][w]; s2 += parts[1][w]; }
    cham[g] = (s1 + s2) / (float)kNPG;
  }
}

__global__ void loss_kernel(const float* __restrict__ cham,
                            float* __restrict__ out) {
  if (blockIdx.x == 0 && threadIdx.x == 0) {
    float s = 0.0f;
    for (int g = 0; g < kB; ++g) s += cham[g];
    out[0] = s / (float)kB;
  }
}

// ------------------------------ launch -------------------------------------
extern "C" void kernel_launch(void* const* d_in, const int* in_sizes, int n_in,
                              void* d_out, int out_size, void* d_ws, size_t ws_size,
                              hipStream_t stream) {
  const float* x   = (const float*)d_in[0];
  const int*   ei  = (const int*)d_in[1];
  const float* pos = (const float*)d_in[2];
  const float* W1  = (const float*)d_in[4];
  const float* b1  = (const float*)d_in[5];
  const float* W2  = (const float*)d_in[6];
  const float* b2  = (const float*)d_in[7];
  const float* W3  = (const float*)d_in[8];
  const float* b3  = (const float*)d_in[9];
  const float* Wt  = (const float*)d_in[10];
  const float* bt  = (const float*)d_in[11];
  const int* row = ei;
  const int* col = ei + kE;

  float* out_h       = (float*)d_out;                       // N*OUT
  float* out_aligned = out_h + (size_t)kN * kOUT;           // N*3
  float* out_loss    = out_aligned + (size_t)kN * 3;        // 1

  float* ws   = (float*)d_ws;
  float* bufA = ws;                          // N*HID  (xw)
  float* bufB = bufA + (size_t)kN * kHID;    // N*HID  (scatter acc)
  float* bufC = bufB + (size_t)kN * kHID;    // N*HID  (hidden)
  float* dinv = bufC + (size_t)kN * kHID;    // N      (deg -> dinv in place)
  float* tpos = dinv + kN;                   // N*3
  float* cham = tpos + (size_t)kN * 3;       // B

  // --- degree normalization (once) ---
  deg_init_kernel<<<kN / 256, 256, 0, stream>>>(dinv, kN);
  deg_edge_kernel<<<kE / 256, 256, 0, stream>>>(col, dinv, kE);
  dinv_kernel<<<kN / 256, 256, 0, stream>>>(dinv, kN);

  // --- layer 1: x[N,128] @ W1[128,256] -> relu GCN ---
  {
    int waves = (kN / 16) * (kHID / 32);
    gemm_wmma_bf16x3<kIN><<<waves / 8, 256, 0, stream>>>(x, W1, bufA, kN, kHID);
    (void)hipMemsetAsync(bufB, 0, (size_t)kN * kHID * sizeof(float), stream);
    int threads = kE * (kHID / 4);
    gcn_scatter_kernel<<<threads / 256, 256, 0, stream>>>(bufA, bufB, row, col,
                                                          dinv, kE, kHID);
    gcn_finalize_kernel<<<(kN * kHID) / 256, 256, 0, stream>>>(
        bufA, bufB, b1, dinv, bufC, kN, kHID, 1);
  }
  // --- layer 2: h1[N,256] @ W2[256,256] -> relu GCN ---
  {
    int waves = (kN / 16) * (kHID / 32);
    gemm_wmma_bf16x3<kHID><<<waves / 8, 256, 0, stream>>>(bufC, W2, bufA, kN,
                                                          kHID);
    (void)hipMemsetAsync(bufB, 0, (size_t)kN * kHID * sizeof(float), stream);
    int threads = kE * (kHID / 4);
    gcn_scatter_kernel<<<threads / 256, 256, 0, stream>>>(bufA, bufB, row, col,
                                                          dinv, kE, kHID);
    gcn_finalize_kernel<<<(kN * kHID) / 256, 256, 0, stream>>>(
        bufA, bufB, b2, dinv, bufC, kN, kHID, 1);
  }
  // --- layer 3: h2[N,256] @ W3[256,128] -> GCN (no relu) -> d_out h ---
  {
    int waves = (kN / 16) * (kOUT / 32);
    gemm_wmma_bf16x3<kHID><<<waves / 8, 256, 0, stream>>>(bufC, W3, bufA, kN,
                                                          kOUT);
    (void)hipMemsetAsync(bufB, 0, (size_t)kN * kOUT * sizeof(float), stream);
    int threads = kE * (kOUT / 4);
    gcn_scatter_kernel<<<threads / 256, 256, 0, stream>>>(bufA, bufB, row, col,
                                                          dinv, kE, kOUT);
    gcn_finalize_kernel<<<(kN * kOUT) / 256, 256, 0, stream>>>(
        bufA, bufB, b3, dinv, out_h, kN, kOUT, 0);
  }
  // --- translation head: tpos = pos + h @ Wt + bt ---
  head_kernel<<<(kN * 3 + 255) / 256, 256, 0, stream>>>(out_h, Wt, bt, pos,
                                                        tpos, kN);
  // --- fused ICP (10 iters) + chamfer, one workgroup per graph ---
  icp_chamfer_kernel<<<kB, kIcpThreads, 0, stream>>>(tpos, pos, out_aligned,
                                                     cham);
  loss_kernel<<<1, 32, 0, stream>>>(cham, out_loss);
}
